// NeuralNetwork_75479755260510
// MI455X (gfx1250) — matrix operand order repaired and checked by reading, not compiled, not run on hardware
//
#include <hip/hip_runtime.h>
#include <stdint.h>

// ---------------------------------------------------------------------------
// 2-layer GCN for MI455X (gfx1250, wave32, WMMA).
//  * feature GEMMs: v_wmma_f32_16x16x32_bf16 on pre-converted bf16 activations
//  * W1 staged into LDS via global_load_async_to_lds_b128 (ASYNCcnt)
//  * sparse aggregation: f32 L2 atomics (footprints fit in 192MB L2)
// ---------------------------------------------------------------------------

typedef __attribute__((ext_vector_type(16))) __bf16        v16bf;
typedef __attribute__((ext_vector_type(8)))  float         v8f;
typedef __attribute__((ext_vector_type(8)))  unsigned int  v8u;

#define GCN_FEAT 256
#define GCN_HID  64
#define GCN_OUT  40
#define GCN_OUTP 48   // OUT padded to a multiple of 16 for WMMA tiles

// f32 -> bf16, round-to-nearest-even (only used in streaming/setup code)
__device__ __forceinline__ unsigned short f2bf(float f) {
  unsigned u = __builtin_bit_cast(unsigned, f);
  u += 0x7FFFu + ((u >> 16) & 1u);
  return (unsigned short)(u >> 16);
}

__device__ __forceinline__ v16bf pack8(const uint4& a0, const uint4& a1) {
  v8u v = {a0.x, a0.y, a0.z, a0.w, a1.x, a1.y, a1.z, a1.w};
  return __builtin_bit_cast(v16bf, v);
}

// 16-bit A fragment (16x32, MxK), contiguous K order (same as B below):
//   lanes 0-15 : M=lane,    elems 0..15 = K k0+0..15
//   lanes 16-31: M=lane-16, elems 0..15 = K k0+16..31
// Row already bf16 -> two contiguous 16B loads, zero conversion VALU.
__device__ __forceinline__ v16bf a_frag_bf16(const unsigned short* __restrict__ row,
                                             int k0, bool hi) {
  const unsigned short* p0 = row + k0 + (hi ? 16 : 0);
  const uint4* q0 = reinterpret_cast<const uint4*>(p0);
  const uint4* q1 = reinterpret_cast<const uint4*>(p0 + 8);
  return pack8(q0[0], q1[0]);
}

// 16-bit B fragment (32x16, KxN): lane n holds column n; lanes 0-15 carry
// K k0..k0+15, lanes 16-31 carry K k0+16..31 -> 16 contiguous bf16 from a
// K-major (transposed-weight) LDS row.
__device__ __forceinline__ v16bf b_frag_lds(const unsigned short* p) {
  const uint4* q = reinterpret_cast<const uint4*>(p);
  return pack8(q[0], q[1]);
}

__device__ __forceinline__ v8f wmma_bf16(v16bf a, v16bf b, v8f c) {
  return __builtin_amdgcn_wmma_f32_16x16x32_bf16(false, a, false, b,
                                                 (short)0, c, false, false);
}

// ------------------------- x (f32) -> bf16, streaming -----------------------
__global__ void __launch_bounds__(256)
gcn_f32_to_bf16(const float* __restrict__ in, unsigned short* __restrict__ out,
                long long n8) {           // n8 = elements / 8
  long long i = (long long)blockIdx.x * blockDim.x + threadIdx.x;
  if (i >= n8) return;
  const float4* p = reinterpret_cast<const float4*>(in) + i * 2;
  float4 a = p[0], b = p[1];
  uint4 o;
  o.x = f2bf(a.x) | ((unsigned)f2bf(a.y) << 16);
  o.y = f2bf(a.z) | ((unsigned)f2bf(a.w) << 16);
  o.z = f2bf(b.x) | ((unsigned)f2bf(b.y) << 16);
  o.w = f2bf(b.z) | ((unsigned)f2bf(b.w) << 16);
  reinterpret_cast<uint4*>(out)[i] = o;
}

// ------------------------- GEMM 1: h1 = xb @ W1 [N,256]x[256,64] ------------
__global__ void __launch_bounds__(256)
gcn_gemm1(const unsigned short* __restrict__ xb, const float* __restrict__ W1,
          float* __restrict__ h1, int nNodes) {
  const int WS = GCN_FEAT + 16;                 // padded K stride (bank-spread)
  __shared__ float          Wsrc[GCN_FEAT * GCN_HID];  // raw f32 staging (64KB)
  __shared__ unsigned short Wt[GCN_HID * WS];          // W1^T as bf16: [n][k]

  // Stage W1 (f32, 64KB) into LDS with CDNA5 async copy (ASYNCcnt-tracked).
  // Low 32 bits of a flat LDS-aperture address == wave LDS byte offset.
  {
    unsigned ldsBase = (unsigned)(uintptr_t)(&Wsrc[0]);
    for (int i = threadIdx.x; i < (GCN_FEAT * GCN_HID) / 4; i += 256) {
      unsigned laddr = ldsBase + (unsigned)(i * 16);
      const float* gaddr = W1 + i * 4;
      asm volatile("global_load_async_to_lds_b128 %0, %1, off"
                   :: "v"(laddr), "v"(gaddr) : "memory");
    }
    asm volatile("s_wait_asynccnt 0x0" ::: "memory");
  }
  __syncthreads();
  // convert + transpose inside LDS: Wt[n][k] = bf16(W1[k][n])
  for (int i = threadIdx.x; i < GCN_FEAT * GCN_HID; i += 256) {
    int k = i >> 6, n = i & 63;                 // W1 row-major [256][64]
    Wt[n * WS + k] = f2bf(Wsrc[i]);
  }
  __syncthreads();

  const int wave  = threadIdx.x >> 5;
  const int lane  = threadIdx.x & 31;
  const int laneM = lane & 15;
  const bool hi   = lane >= 16;
  const int row0  = blockIdx.x * 128 + wave * 16;   // 8 waves x 16 rows
  int arow = row0 + laneM;
  if (arow >= nNodes) arow = nNodes - 1;            // clamp (store is guarded)
  const unsigned short* rowPtr = xb + (size_t)arow * GCN_FEAT;
  __builtin_prefetch(rowPtr, 0, 3);                 // global_prefetch_b8

  v8f acc0 = {}, acc1 = {}, acc2 = {}, acc3 = {};
#pragma unroll
  for (int k0 = 0; k0 < GCN_FEAT; k0 += 32) {       // 8 K-slabs -> 32 WMMAs
    v16bf a = a_frag_bf16(rowPtr, k0, hi);
    const int kb = k0 + (hi ? 16 : 0);
    v16bf b0 = b_frag_lds(&Wt[(laneM)      * WS + kb]);
    v16bf b1 = b_frag_lds(&Wt[(16 + laneM) * WS + kb]);
    v16bf b2 = b_frag_lds(&Wt[(32 + laneM) * WS + kb]);
    v16bf b3 = b_frag_lds(&Wt[(48 + laneM) * WS + kb]);
    acc0 = wmma_bf16(a, b0, acc0);
    acc1 = wmma_bf16(a, b1, acc1);
    acc2 = wmma_bf16(a, b2, acc2);
    acc3 = wmma_bf16(a, b3, acc3);
  }
  // C layout: VGPR r -> M = r (lanes 0-15) or 8+r (lanes 16-31), N = laneM
#pragma unroll
  for (int r = 0; r < 8; ++r) {
    int row = row0 + r + (hi ? 8 : 0);
    if (row < nNodes) {
      float* o = h1 + (size_t)row * GCN_HID + laneM;
      o[0]  = acc0[r];
      o[16] = acc1[r];
      o[32] = acc2[r];
      o[48] = acc3[r];
    }
  }
}

// ------------------------- GEMM 2: h2 = hb @ W2 [N,64]x[64,40->48] ----------
__global__ void __launch_bounds__(256)
gcn_gemm2(const unsigned short* __restrict__ hb, const float* __restrict__ W2,
          float* __restrict__ h2, int nNodes) {
  const int WS = GCN_HID + 8;
  __shared__ unsigned short Wt[GCN_OUTP * WS];      // W2^T bf16, zero-padded N
  for (int i = threadIdx.x; i < GCN_OUTP * GCN_HID; i += 256) {
    int n = i % GCN_OUTP, k = i / GCN_OUTP;
    Wt[n * WS + k] = (n < GCN_OUT) ? f2bf(W2[k * GCN_OUT + n])
                                   : (unsigned short)0;
  }
  __syncthreads();

  const int wave  = threadIdx.x >> 5;
  const int lane  = threadIdx.x & 31;
  const int laneM = lane & 15;
  const bool hi   = lane >= 16;
  const int row0  = blockIdx.x * 128 + wave * 16;
  int arow = row0 + laneM;
  if (arow >= nNodes) arow = nNodes - 1;
  const unsigned short* rowPtr = hb + (size_t)arow * GCN_HID;

  v8f acc0 = {}, acc1 = {}, acc2 = {};
#pragma unroll
  for (int k0 = 0; k0 < GCN_HID; k0 += 32) {        // 2 K-slabs -> 6 WMMAs
    v16bf a = a_frag_bf16(rowPtr, k0, hi);
    const int kb = k0 + (hi ? 16 : 0);
    v16bf b0 = b_frag_lds(&Wt[(laneM)      * WS + kb]);
    v16bf b1 = b_frag_lds(&Wt[(16 + laneM) * WS + kb]);
    v16bf b2 = b_frag_lds(&Wt[(32 + laneM) * WS + kb]);
    acc0 = wmma_bf16(a, b0, acc0);
    acc1 = wmma_bf16(a, b1, acc1);
    acc2 = wmma_bf16(a, b2, acc2);
  }
#pragma unroll
  for (int r = 0; r < 8; ++r) {
    int row = row0 + r + (hi ? 8 : 0);
    if (row < nNodes) {
      float* o = h2 + (size_t)row * GCN_OUT;        // stride 40, cols 40..47 dropped
      o[laneM]      = acc0[r];
      o[16 + laneM] = acc1[r];
      if (laneM < 8) o[32 + laneM] = acc2[r];
    }
  }
}

// ------------------------- degree / norm ------------------------------------
__global__ void gcn_deg_init(float* __restrict__ deg, int n) {
  int i = blockIdx.x * blockDim.x + threadIdx.x;
  if (i < n) deg[i] = 1.0f;                         // self loop
}
__global__ void gcn_deg_count(const long long* __restrict__ dst,
                              float* __restrict__ deg, int nEdges) {
  int i = blockIdx.x * blockDim.x + threadIdx.x;
  if (i < nEdges) atomicAdd(&deg[(int)dst[i]], 1.0f);
}
__global__ void gcn_deg_rsqrt(float* __restrict__ deg, int n) {
  int i = blockIdx.x * blockDim.x + threadIdx.x;
  if (i < n) deg[i] = rsqrtf(deg[i]);
}

// ------------------------- zero fill ----------------------------------------
__global__ void gcn_zero(float* __restrict__ p, long long n) {
  long long i = (long long)blockIdx.x * blockDim.x + threadIdx.x;
  if (i < n) p[i] = 0.0f;
}

// ------------------------- gather * norm -> scatter-add ---------------------
// virtual edges e >= nEdges are the self loops (src = dst = e - nEdges)
template <int F>
__global__ void __launch_bounds__(256)
gcn_aggregate(const float* __restrict__ h, const long long* __restrict__ src,
              const long long* __restrict__ dst, const float* __restrict__ dinv,
              float* __restrict__ out, int nEdges, int nNodes) {
  constexpr int vecs = F / 4;
  long long tid = (long long)blockIdx.x * blockDim.x + threadIdx.x;
  long long total = (long long)(nEdges + nNodes) * vecs;
  if (tid >= total) return;
  int e = (int)(tid / vecs);
  int q = (int)(tid % vecs) << 2;
  int s, d;
  if (e < nEdges) { s = (int)src[e]; d = (int)dst[e]; }
  else            { s = d = e - nEdges; }
  float nm = dinv[s] * dinv[d];
  const float4 v = *reinterpret_cast<const float4*>(h + (size_t)s * F + q);
  float* o = out + (size_t)d * F + q;
  atomicAdd(o + 0, v.x * nm);
  atomicAdd(o + 1, v.y * nm);
  atomicAdd(o + 2, v.z * nm);
  atomicAdd(o + 3, v.w * nm);
}

// ------------------------- relu(agg + b1) -> bf16 ---------------------------
__global__ void __launch_bounds__(256)
gcn_relu_bias_bf16(const float* __restrict__ agg, const float* __restrict__ b1,
                   unsigned short* __restrict__ hb, long long total) {
  long long i = (long long)blockIdx.x * blockDim.x + threadIdx.x;
  if (i >= total) return;
  float v = agg[i] + b1[(int)(i & (GCN_HID - 1))];
  hb[i] = f2bf(fmaxf(v, 0.0f));
}

// ------------------------- bias + log_softmax (in place) --------------------
__global__ void __launch_bounds__(256)
gcn_bias_logsoftmax(float* __restrict__ out, const float* __restrict__ b2,
                    int nNodes) {
  int row = blockIdx.x * blockDim.x + threadIdx.x;
  if (row >= nNodes) return;
  float* p = out + (size_t)row * GCN_OUT;
  float v[GCN_OUT];
  float m = -3.402823466e38f;
#pragma unroll
  for (int j = 0; j < GCN_OUT; ++j) { v[j] = p[j] + b2[j]; m = fmaxf(m, v[j]); }
  float s = 0.0f;
#pragma unroll
  for (int j = 0; j < GCN_OUT; ++j) s += __expf(v[j] - m);
  float ls = __logf(s) + m;
#pragma unroll
  for (int j = 0; j < GCN_OUT; ++j) p[j] = v[j] - ls;
}

// ---------------------------------------------------------------------------
extern "C" void kernel_launch(void* const* d_in, const int* in_sizes, int n_in,
                              void* d_out, int out_size, void* d_ws, size_t ws_size,
                              hipStream_t stream) {
  const float*     x    = (const float*)d_in[0];
  const long long* edge = (const long long*)d_in[1];   // int64 [2, E]
  const float*     W1   = (const float*)d_in[2];
  const float*     b1   = (const float*)d_in[3];
  const float*     W2   = (const float*)d_in[4];
  const float*     b2   = (const float*)d_in[5];
  float*           out  = (float*)d_out;

  const int N = in_sizes[0] / GCN_FEAT;
  const int E = in_sizes[1] / 2;
  const long long* src = edge;
  const long long* dst = edge + E;

  // scratch partition (~45 MB with aliasing):
  //   regionA (25.6MB): x_bf16 during GEMM1, then agg1 (12.8MB) + h2 (8MB)
  char*  base = (char*)d_ws;
  size_t off  = 0;
  auto carve = [&](size_t bytes) -> char* {
    char* p = base + off;
    off += (bytes + 255) & ~(size_t)255;
    return p;
  };
  float*          dinv    = (float*)carve((size_t)N * 4);
  char*           regionA = carve((size_t)N * GCN_FEAT * 2);
  unsigned short* xb      = (unsigned short*)regionA;                      // live: convert..gemm1
  float*          agg1    = (float*)regionA;                               // live: after gemm1
  float*          h2      = (float*)(regionA + (size_t)N * GCN_HID * 4);   // live: after gemm1
  float*          h1      = (float*)carve((size_t)N * GCN_HID * 4);
  unsigned short* hb      = (unsigned short*)carve((size_t)N * GCN_HID * 2);
  (void)n_in; (void)ws_size; (void)out_size;

  const int B = 256;

  // symmetric normalization coefficients
  gcn_deg_init <<<(N + B - 1) / B, B, 0, stream>>>(dinv, N);
  gcn_deg_count<<<(E + B - 1) / B, B, 0, stream>>>(dst, dinv, E);
  gcn_deg_rsqrt<<<(N + B - 1) / B, B, 0, stream>>>(dinv, N);

  // d_out is poisoned, not re-zeroed between replays
  long long n1 = (long long)N * GCN_HID;
  long long n2 = (long long)N * GCN_OUT;
  gcn_zero<<<(int)((n2 + B - 1) / B), B, 0, stream>>>(out, n2);

  // activations f32 -> bf16 (streaming; conversion VALU hidden under HBM BW)
  long long n8 = (long long)N * GCN_FEAT / 8;
  gcn_f32_to_bf16<<<(int)((n8 + B - 1) / B), B, 0, stream>>>(x, xb, n8);

  // layer 1: WMMA GEMM -> (zero agg1 AFTER gemm1: it aliases xb) -> aggregate
  gcn_gemm1<<<(N + 127) / 128, B, 0, stream>>>(xb, W1, h1, N);
  gcn_zero<<<(int)((n1 + B - 1) / B), B, 0, stream>>>(agg1, n1);
  long long t1 = (long long)(E + N) * (GCN_HID / 4);
  gcn_aggregate<GCN_HID><<<(int)((t1 + B - 1) / B), B, 0, stream>>>(
      h1, src, dst, dinv, agg1, E, N);
  gcn_relu_bias_bf16<<<(int)((n1 + B - 1) / B), B, 0, stream>>>(agg1, b1, hb, n1);

  // layer 2: WMMA GEMM -> aggregate into d_out -> bias + log_softmax
  gcn_gemm2<<<(N + 127) / 128, B, 0, stream>>>(hb, W2, h2, N);
  long long t2 = (long long)(E + N) * (GCN_OUT / 4);
  gcn_aggregate<GCN_OUT><<<(int)((t2 + B - 1) / B), B, 0, stream>>>(
      h2, src, dst, dinv, out, E, N);
  gcn_bias_logsoftmax<<<(N + B - 1) / B, B, 0, stream>>>(out, b2, N);
}
